// NearestEmbedEMA_24352464568746
// MI455X (gfx1250) — matrix operand order repaired and checked by reading, not compiled, not run on hardware
//
#include <hip/hip_runtime.h>

typedef __attribute__((ext_vector_type(16))) __bf16 v16bf;
typedef __attribute__((ext_vector_type(8)))  float  v8f;
typedef __attribute__((ext_vector_type(4)))  int    v4i;

#define B_    64
#define D_    256
#define HW_   1024
#define M_    65536
#define N_    2048
#define MTILE 256          // pixels per block (8 waves x 2 subtiles x 16 rows)
#define NTILE 128          // codebook columns per LDS tile
#define KPAD  264          // 256 + 8 ushort pad -> 528B rows, spreads LDS banks
#define WBUF  (NTILE * KPAD)
#define NT_   (N_ / NTILE)

#if defined(__has_builtin)
#if __has_builtin(__builtin_amdgcn_global_load_async_to_lds_b128)
#define HAVE_ASYNC_LDS 1
#endif
#endif

#ifdef HAVE_ASYNC_LDS
typedef __attribute__((address_space(1))) v4i* gvec_ptr;   // global b128
typedef __attribute__((address_space(3))) v4i* lvec_ptr;   // LDS b128
#endif

__device__ __forceinline__ unsigned short f2bf(float f) {
  unsigned u = __float_as_uint(f);
  unsigned r = u + 0x7FFFu + ((u >> 16) & 1u);   // round-to-nearest-even
  return (unsigned short)(r >> 16);
}

// ---------------- prep: weight -> bf16 transposed + exact fp32 w2 ----------------
__global__ void prep_weight(const float* __restrict__ w,
                            unsigned short* __restrict__ wT,
                            float* __restrict__ w2) {
  int n = blockIdx.x * blockDim.x + threadIdx.x;      // 0..2047, coalesced over n
  float s = 0.f;
  for (int d = 0; d < D_; ++d) {
    float v = w[d * N_ + n];
    s += v * v;
    wT[n * D_ + d] = f2bf(v);
  }
  w2[n] = s;
}

// fill one NTILE x D_ bf16 weight tile into LDS (async memory->LDS when available)
__device__ __forceinline__ void fill_wtile(const unsigned short* __restrict__ wT,
                                           unsigned short* dstbuf, int tn, int tid) {
  const uint4* src = (const uint4*)(wT + (size_t)tn * NTILE * D_);
  for (int e = tid; e < NTILE * D_ / 8; e += 256) {
    int col = e >> 5;            // 32 x 16B chunks per column
    int kk  = (e & 31) * 8;
#ifdef HAVE_ASYNC_LDS
    __builtin_amdgcn_global_load_async_to_lds_b128(
        (gvec_ptr)(src + e),
        (lvec_ptr)(dstbuf + col * KPAD + kk),
        0, 0);
#else
    *(uint4*)(dstbuf + col * KPAD + kk) = src[e];
#endif
  }
}

__device__ __forceinline__ void wait_async_fill() {
#ifdef HAVE_ASYNC_LDS
  asm volatile("s_wait_asynccnt 0x0" ::: "memory");
#endif
}

// ---------------- fused GEMM + argmin over codes ----------------
__launch_bounds__(256)
__global__ void vq_argmin(const float* __restrict__ x,
                          const unsigned short* __restrict__ wT,
                          const float* __restrict__ w2,
                          int* __restrict__ amin) {
  __shared__ unsigned short xs[MTILE * KPAD];   // 132 KB bf16 x tile [pixel][d]
  __shared__ unsigned short wls[2 * WBUF];      // 2 x 66 KB double-buffered w tile
  __shared__ float w2s[N_];                     // 8 KB

  const int tid = threadIdx.x;
  const int pixelBase = blockIdx.x * MTILE;     // 256 | 1024: tile stays in one b
  const int b = pixelBase >> 10;
  const int hwBase = pixelBase & 1023;

  // prefetch first weight tile (overlaps with the x-tile conversion below)
  fill_wtile(wT, wls, 0, tid);

  // load x tile -> bf16 LDS (thread = pixel; coalesced over hw for each d)
  {
    const float* xb = x + (size_t)b * D_ * HW_ + hwBase + tid;
    for (int d = 0; d < D_; ++d)
      xs[tid * KPAD + d] = f2bf(xb[(size_t)d * HW_]);
  }
  for (int i = tid; i < N_; i += 256) w2s[i] = w2[i];

  wait_async_fill();
  __syncthreads();

  const int lane = tid & 31;
  const int wv   = tid >> 5;                    // wave 0..7 owns 32 pixels
  const int half = lane >> 4;
  const int l16  = lane & 15;
  const int rowBase = wv * 32;

  // Two resident A-fragment sets (ISA 16-bit A 16x32 layout:
  // lanes 0-15: K {0-7,16-23}; lanes 16-31: K {8-15,24-31}; M = lane&15)
  v16bf a0[8], a1[8];
  {
    const unsigned short* xr0 = &xs[(rowBase + l16) * KPAD];
    const unsigned short* xr1 = &xs[(rowBase + 16 + l16) * KPAD];
    #pragma unroll
    for (int c = 0; c < 8; ++c) {
      union { uint4 u[2]; v16bf v; } f;
      f.u[0] = *(const uint4*)(xr0 + c * 32 + 8 * half);
      f.u[1] = *(const uint4*)(xr0 + c * 32 + 16 + 8 * half);
      a0[c] = f.v;
      f.u[0] = *(const uint4*)(xr1 + c * 32 + 8 * half);
      f.u[1] = *(const uint4*)(xr1 + c * 32 + 16 + 8 * half);
      a1[c] = f.v;
    }
  }

  float minv0[8], minv1[8];
  int   mini0[8], mini1[8];
  #pragma unroll
  for (int r = 0; r < 8; ++r) {
    minv0[r] = 3.4e38f; mini0[r] = 0;
    minv1[r] = 3.4e38f; mini1[r] = 0;
  }

  for (int tn = 0; tn < NT_; ++tn) {
    const unsigned short* cur = &wls[(tn & 1) * WBUF];
    if (tn + 1 < NT_)                            // async prefetch of next tile
      fill_wtile(wT, &wls[((tn + 1) & 1) * WBUF], tn + 1, tid);

    for (int j = 0; j < NTILE / 16; ++j) {
      v8f acc0 = {}, acc1 = {};
      // B 32x16 layout: N = lane&15; lanes 0-15 hold K {0-7,16-23}, lanes 16-31 K {8-15,24-31}
      const unsigned short* wr = &cur[(j * 16 + l16) * KPAD + 8 * half];
      #pragma unroll
      for (int c = 0; c < 8; ++c) {
        union { uint4 u[2]; v16bf v; } f;
        f.u[0] = *(const uint4*)(wr + c * 32);
        f.u[1] = *(const uint4*)(wr + c * 32 + 16);
        acc0 = __builtin_amdgcn_wmma_f32_16x16x32_bf16(false, a0[c], false, f.v,
                                                       (short)0, acc0, false, false);
        acc1 = __builtin_amdgcn_wmma_f32_16x16x32_bf16(false, a1[c], false, f.v,
                                                       (short)0, acc1, false, false);
      }
      // C layout: VGPR r, lane l -> row r + 8*(l>=16), col l&15
      int   code = tn * NTILE + j * 16 + l16;
      float w2v  = w2s[code];
      #pragma unroll
      for (int r = 0; r < 8; ++r) {
        float s0 = w2v - 2.0f * acc0[r];        // argmin proxy: w^2 - 2 x.w
        if (s0 < minv0[r]) { minv0[r] = s0; mini0[r] = code; }
        float s1 = w2v - 2.0f * acc1[r];
        if (s1 < minv1[r]) { minv1[r] = s1; mini1[r] = code; }
      }
    }

    wait_async_fill();                           // drain prefetch, then swap
    __syncthreads();
  }

  // reduce over the 16 columns held by each 16-lane group
  #pragma unroll
  for (int r = 0; r < 8; ++r) {
    #pragma unroll
    for (int m = 8; m >= 1; m >>= 1) {
      float ov0 = __shfl_xor(minv0[r], m, 32);
      int   oi0 = __shfl_xor(mini0[r], m, 32);
      if (ov0 < minv0[r] || (ov0 == minv0[r] && oi0 < mini0[r])) { minv0[r] = ov0; mini0[r] = oi0; }
      float ov1 = __shfl_xor(minv1[r], m, 32);
      int   oi1 = __shfl_xor(mini1[r], m, 32);
      if (ov1 < minv1[r] || (ov1 == minv1[r] && oi1 < mini1[r])) { minv1[r] = ov1; mini1[r] = oi1; }
    }
  }
  if (l16 == 0) {
    #pragma unroll
    for (int r = 0; r < 8; ++r) {
      int p0 = pixelBase + rowBase + half * 8 + r;
      amin[p0]      = mini0[r];
      amin[p0 + 16] = mini1[r];
    }
  }
}

// ---------------- gather: out[b,d,h,w] = weight[d, amin[b,h,w]] ----------------
__global__ void vq_gather(const float* __restrict__ w,
                          const int* __restrict__ amin,
                          float* __restrict__ out) {
  int o  = blockIdx.x * 256 + threadIdx.x;      // linear (b,d,hw) index
  int hw = o & 1023;
  int bd = o >> 10;
  int d  = bd & 255;
  int b  = bd >> 8;
  int idx = amin[b * HW_ + hw];
  out[o] = w[d * N_ + idx];                     // weight row is L2-resident
}

extern "C" void kernel_launch(void* const* d_in, const int* in_sizes, int n_in,
                              void* d_out, int out_size, void* d_ws, size_t ws_size,
                              hipStream_t stream) {
  const float* x = (const float*)d_in[0];
  const float* w = (const float*)d_in[1];

  float* out  = (float*)d_out;
  int*   amin = (int*)d_out + (size_t)M_ * D_;  // second tuple output region

  unsigned short* wT = (unsigned short*)d_ws;                        // 1 MB
  float*          w2 = (float*)((char*)d_ws + (size_t)N_ * D_ * 2);  // 8 KB

  prep_weight<<<N_ / 256, 256, 0, stream>>>(w, wT, w2);
  vq_argmin  <<<M_ / MTILE, 256, 0, stream>>>(x, wT, w2, amin);
  vq_gather  <<<(M_ * D_) / 256, 256, 0, stream>>>(w, amin, out);
}
